// HAT_30124900614681
// MI455X (gfx1250) — compile-verified, matrix operand order checked by reading, not run on hardware
//
#include <hip/hip_runtime.h>
#include <hip/hip_bf16.h>

// Problem constants (from reference): N=20000, DEG=32, D=128, L=2
#define N_NODES 20000
#define DEG     32
#define DIM     128
#define QKV_DIM 384   // 3*D
#define NLAYERS 2

typedef __bf16 bf16x16 __attribute__((ext_vector_type(16)));
typedef __bf16 bf16x8  __attribute__((ext_vector_type(8)));
typedef __bf16 bf16x4  __attribute__((ext_vector_type(4)));
typedef float  f32x8   __attribute__((ext_vector_type(8)));
typedef float  f32x4   __attribute__((ext_vector_type(4)));

__device__ __forceinline__ float wave_sum(float v) {
#pragma unroll
  for (int off = 16; off > 0; off >>= 1) v += __shfl_xor(v, off, 32);
  return v;
}
__device__ __forceinline__ float wave_max(float v) {
#pragma unroll
  for (int off = 16; off > 0; off >>= 1) v = fmaxf(v, __shfl_xor(v, off, 32));
  return v;
}

// Load one 16x32 bf16 WMMA fragment half-row for this lane.
// Per CDNA5 ISA 7.12.2: lane = idx + 16*h, element e -> K = (e<8?0:16)+8h+(e&7),
// i.e. two contiguous 16-byte chunks at +8h and +16+8h.
__device__ __forceinline__ bf16x16 load_frag(const __bf16* p) {
  bf16x8 lo = *(const bf16x8*)(p);
  bf16x8 hi = *(const bf16x8*)(p + 16);
  bf16x16 f;
#pragma unroll
  for (int i = 0; i < 8; ++i) { f[i] = lo[i]; f[i + 8] = hi[i]; }
  return f;
}

// ---------------------------------------------------------------------------
// Convert f32 weights -> bf16 (done once per launch; tiny)
// ---------------------------------------------------------------------------
__global__ void cvt_w_kernel(const float* __restrict__ in_w,
                             const float* __restrict__ out_w,
                             __bf16* __restrict__ win_bf,
                             __bf16* __restrict__ wout_bf) {
  const int n_in  = NLAYERS * QKV_DIM * DIM;  // 98304
  const int n_out = NLAYERS * DIM * DIM;      // 32768
  int idx = blockIdx.x * blockDim.x + threadIdx.x;
  if (idx < n_in) win_bf[idx] = (__bf16)in_w[idx];
  int j = idx - n_in;
  if (j >= 0 && j < n_out) wout_bf[j] = (__bf16)out_w[j];
}

// ---------------------------------------------------------------------------
// t = logmap0(x); cast to bf16.  One wave per row (lane owns 4 dims).
// ---------------------------------------------------------------------------
__global__ void logmap_cast_kernel(const float* __restrict__ x,
                                   __bf16* __restrict__ t,
                                   const float* __restrict__ curv) {
  int wave = threadIdx.x >> 5;
  int lane = threadIdx.x & 31;
  int row  = blockIdx.x * 8 + wave;
  if (row >= N_NODES) return;
  f32x4 v = *(const f32x4*)(x + (size_t)row * DIM + lane * 4);
  float ss = v[0]*v[0] + v[1]*v[1] + v[2]*v[2] + v[3]*v[3];
  ss = wave_sum(ss);
  float norm = fmaxf(sqrtf(ss), 1e-7f);
  float sc   = sqrtf(fabsf(curv[0]));
  float arg  = fminf(sc * norm, 1.0f - 1e-5f);
  float fac  = atanhf(arg) / (sc * norm);
  bf16x4 o;
  o[0] = (__bf16)(v[0] * fac); o[1] = (__bf16)(v[1] * fac);
  o[2] = (__bf16)(v[2] * fac); o[3] = (__bf16)(v[3] * fac);
  *(bf16x4*)(t + (size_t)row * DIM + lane * 4) = o;
}

// ---------------------------------------------------------------------------
// x = expmap0(g).  Same structure, f32 out.
// ---------------------------------------------------------------------------
__global__ void expmap_kernel(const float* __restrict__ g,
                              float* __restrict__ xo,
                              const float* __restrict__ curv) {
  int wave = threadIdx.x >> 5;
  int lane = threadIdx.x & 31;
  int row  = blockIdx.x * 8 + wave;
  if (row >= N_NODES) return;
  f32x4 v = *(const f32x4*)(g + (size_t)row * DIM + lane * 4);
  float ss = v[0]*v[0] + v[1]*v[1] + v[2]*v[2] + v[3]*v[3];
  ss = wave_sum(ss);
  float norm = fmaxf(sqrtf(ss), 1e-7f);
  float sc   = sqrtf(fabsf(curv[0]));
  float fac  = tanhf(sc * norm) / (sc * norm);
  f32x4 o;
  o[0] = v[0] * fac; o[1] = v[1] * fac; o[2] = v[2] * fac; o[3] = v[3] * fac;
  *(f32x4*)(xo + (size_t)row * DIM + lane * 4) = o;
}

// ---------------------------------------------------------------------------
// C[rows x M] = A[rows x 128]_bf16 * W[M x 128]_bf16 ^T + bias, f32 accum.
// One wave per 16x64 C strip: A fragment loaded once per K-step, reused by
// 4 WMMAs against 4 B fragments (2.5 loads per v_wmma instead of 4).
// 16 x v_wmma_f32_16x16x32_bf16 per wave over K=128.
// ---------------------------------------------------------------------------
__global__ void gemm_bf16_kernel(const __bf16* __restrict__ A,
                                 const __bf16* __restrict__ W,
                                 const float* __restrict__ bias,
                                 float* __restrict__ C,
                                 int ldc, int nColQuads, int nTiles) {
  int wave = threadIdx.x >> 5;
  int lane = threadIdx.x & 31;
  int id   = blockIdx.x * 8 + wave;
  if (id >= nTiles) return;
  int rowTile = (id / nColQuads) * 16;
  int colBase = (id % nColQuads) * 64;
  int h = lane >> 4;    // K half-group selector
  int r = lane & 15;    // M index (for A) / N index (for B)

  const __bf16* Arow = A + (size_t)(rowTile + r) * DIM + 8 * h;
  const __bf16* Wrow = W + (size_t)(colBase + r) * DIM + 8 * h;

  f32x8 acc0 = {}, acc1 = {}, acc2 = {}, acc3 = {};
#pragma unroll
  for (int k0 = 0; k0 < DIM; k0 += 32) {
    bf16x16 a  = load_frag(Arow + k0);
    bf16x16 b0 = load_frag(Wrow + k0);
    bf16x16 b1 = load_frag(Wrow + k0 + 16 * DIM);
    bf16x16 b2 = load_frag(Wrow + k0 + 32 * DIM);
    bf16x16 b3 = load_frag(Wrow + k0 + 48 * DIM);
    acc0 = __builtin_amdgcn_wmma_f32_16x16x32_bf16(false, a, false, b0, (short)0, acc0, false, false);
    acc1 = __builtin_amdgcn_wmma_f32_16x16x32_bf16(false, a, false, b1, (short)0, acc1, false, false);
    acc2 = __builtin_amdgcn_wmma_f32_16x16x32_bf16(false, a, false, b2, (short)0, acc2, false, false);
    acc3 = __builtin_amdgcn_wmma_f32_16x16x32_bf16(false, a, false, b3, (short)0, acc3, false, false);
  }

  // D layout: VGPR i holds row rowTile + i + 8*h, column = tile base + (lane&15)
#pragma unroll
  for (int t = 0; t < 4; ++t) {
    f32x8 acc = (t == 0) ? acc0 : (t == 1) ? acc1 : (t == 2) ? acc2 : acc3;
    int col = colBase + 16 * t + r;
    float bv = bias[col];
#pragma unroll
    for (int i = 0; i < 8; ++i) {
      int row = rowTile + i + 8 * h;
      C[(size_t)row * ldc + col] = acc[i] + bv;
    }
  }
}

// ---------------------------------------------------------------------------
// Attention: one wave per node, lane = neighbor (DEG==32==wave32).
// scores -> shfl softmax -> role-swap: lane owns 4 output dims for the
// weighted-V accumulation (all loads 128-bit, L2-resident gathers).
// ---------------------------------------------------------------------------
__global__ void attn_kernel(const float* __restrict__ qkv,
                            const int* __restrict__ nbrs,
                            __bf16* __restrict__ h) {
  int wave = threadIdx.x >> 5;
  int lane = threadIdx.x & 31;
  int node = blockIdx.x * 8 + wave;
  if (node >= N_NODES) return;

  int nb = nbrs[(size_t)node * DEG + lane];
  const float* qrow = qkv + (size_t)node * QKV_DIM;            // q
  const float* krow = qkv + (size_t)nb * QKV_DIM + DIM;        // k[neighbor]

  float s = 0.f;
#pragma unroll 8
  for (int k = 0; k < DIM; k += 4) {
    f32x4 qv = *(const f32x4*)(qrow + k);
    f32x4 kv = *(const f32x4*)(krow + k);
    s += qv[0]*kv[0] + qv[1]*kv[1] + qv[2]*kv[2] + qv[3]*kv[3];
  }
  s *= 0.08838834764831845f;   // 1/sqrt(128)

  float m   = wave_max(s);
  float e   = __expf(s - m);
  float sum = wave_sum(e);
  float a   = e / sum;

  float acc0 = 0.f, acc1 = 0.f, acc2 = 0.f, acc3 = 0.f;
#pragma unroll 4
  for (int j = 0; j < DEG; ++j) {
    float aj  = __shfl(a, j, 32);
    int   nbj = __shfl(nb, j, 32);
    f32x4 vv = *(const f32x4*)(qkv + (size_t)nbj * QKV_DIM + 2 * DIM + lane * 4);
    acc0 += aj * vv[0]; acc1 += aj * vv[1];
    acc2 += aj * vv[2]; acc3 += aj * vv[3];
  }
  bf16x4 o;
  o[0] = (__bf16)acc0; o[1] = (__bf16)acc1;
  o[2] = (__bf16)acc2; o[3] = (__bf16)acc3;
  *(bf16x4*)(h + (size_t)node * DIM + lane * 4) = o;
}

// ---------------------------------------------------------------------------
// Host-side orchestration (graph-capture safe: only kernel launches on stream)
// ---------------------------------------------------------------------------
extern "C" void kernel_launch(void* const* d_in, const int* in_sizes, int n_in,
                              void* d_out, int out_size, void* d_ws, size_t ws_size,
                              hipStream_t stream) {
  const int*   nbrs  = (const int*)d_in[0];
  const float* x0    = (const float*)d_in[1];
  const float* curv  = (const float*)d_in[2];
  const float* in_w  = (const float*)d_in[3];
  const float* in_b  = (const float*)d_in[4];
  const float* out_w = (const float*)d_in[5];
  const float* out_b = (const float*)d_in[6];
  float* xout = (float*)d_out;

  // Workspace layout (256-B aligned; total ~51.5 MB)
  char* ws = (char*)d_ws;
  __bf16* t_bf    = (__bf16*)(ws + 0);            // N*128 bf16  =  5,120,000 B
  float*  qkv     = (float*)(ws + 5120000);       // N*384 f32   = 30,720,000 B
  __bf16* h_bf    = (__bf16*)(ws + 35840000);     // N*128 bf16  =  5,120,000 B
  float*  g       = (float*)(ws + 40960000);      // N*128 f32   = 10,240,000 B
  __bf16* win_bf  = (__bf16*)(ws + 51200000);     // 2*384*128 bf16 = 196,608 B
  __bf16* wout_bf = (__bf16*)(ws + 51396608);     // 2*128*128 bf16 =  65,536 B

  dim3 blk(256);

  cvt_w_kernel<<<512, blk, 0, stream>>>(in_w, out_w, win_bf, wout_bf);

  // qkv GEMM: 1250 row tiles x (384/64)=6 col quads = 7500 waves
  const int qkvTiles = 1250 * 6;
  const int qkvBlocks = (qkvTiles + 7) / 8;
  // out GEMM: 1250 row tiles x (128/64)=2 col quads = 2500 waves
  const int outTiles = 1250 * 2;
  const int outBlocks = (outTiles + 7) / 8;

  for (int l = 0; l < NLAYERS; ++l) {
    const float* xsrc = (l == 0) ? x0 : xout;
    logmap_cast_kernel<<<2500, blk, 0, stream>>>(xsrc, t_bf, curv);
    gemm_bf16_kernel<<<qkvBlocks, blk, 0, stream>>>(
        t_bf, win_bf + (size_t)l * QKV_DIM * DIM, in_b + (size_t)l * QKV_DIM,
        qkv, QKV_DIM, /*nColQuads=*/6, qkvTiles);
    attn_kernel<<<2500, blk, 0, stream>>>(qkv, nbrs, h_bf);
    gemm_bf16_kernel<<<outBlocks, blk, 0, stream>>>(
        h_bf, wout_bf + (size_t)l * DIM * DIM, out_b + (size_t)l * DIM,
        g, DIM, /*nColQuads=*/2, outTiles);
    expmap_kernel<<<2500, blk, 0, stream>>>(g, xout, curv);
  }
}
